// GammaContrastReconLoss_31799937860141
// MI455X (gfx1250) — matrix operand order repaired and checked by reading, not compiled, not run on hardware
//
#include <hip/hip_runtime.h>
#include <hip/hip_bf16.h>
#include <math.h>

typedef __bf16 bf16_t;
typedef __attribute__((ext_vector_type(16))) bf16_t v16bf;
typedef __attribute__((ext_vector_type(8)))  bf16_t v8bf;
typedef __attribute__((ext_vector_type(8)))  float  v8f;

#define B_N   2048
#define D_N   1024
#define S_N   128
#define H_N   512
#define TEMP_INV 10.0f
#define GAMMA 0.5f

// ---- workspace layout (bytes) ----
#define WS_NBF      ((size_t)0)                         // 2048*1024*2  = 4 MB  (bf16 normalized)
#define WS_SIM      ((size_t)(4u*1024*1024))            // 2048*2048*4  = 16 MB (f32 sim)
#define WS_ROWLOSS  (WS_SIM + (size_t)16*1024*1024)     // 2048 * 4
#define WS_ROWVALID (WS_ROWLOSS + (size_t)2048*4)       // 2048 * 4
#define WS_ERRPART  (WS_ROWVALID + (size_t)2048*4)      // 2048 * 4
#define WS_MASKPART (WS_ERRPART + (size_t)2048*4)       // 256 * 4
#define NB_ERR  2048
#define NB_MASK 256

// ---------------------------------------------------------------------------
// K1: per-row L2 normalize latents [2048,1024] f32 -> bf16 workspace
// ---------------------------------------------------------------------------
__global__ void k_normalize_bf16(const float* __restrict__ lat, bf16_t* __restrict__ nbf) {
    __shared__ float sm[256];
    const int row = blockIdx.x;
    const int t = threadIdx.x;
    const float* src = lat + (size_t)row * D_N;

    float ss = 0.f;
    float x[4];
#pragma unroll
    for (int e = 0; e < 4; ++e) {
        x[e] = src[t + e * 256];
        ss += x[e] * x[e];
    }
    sm[t] = ss; __syncthreads();
    for (int off = 128; off > 0; off >>= 1) {
        if (t < off) sm[t] += sm[t + off];
        __syncthreads();
    }
    const float inv = 1.0f / fmaxf(sqrtf(sm[0]), 1e-12f);
    bf16_t* dst = nbf + (size_t)row * D_N;
#pragma unroll
    for (int e = 0; e < 4; ++e) dst[t + e * 256] = (bf16_t)(x[e] * inv);
}

// ---------------------------------------------------------------------------
// K2: Gram matrix sim = (N * N^T) / TEMP via v_wmma_f32_16x16x32_bf16.
// 128x128 tiles of 16x16, one tile per wave, 8 waves (256 thr) per block.
// A layout (16-bit A 16x32, ISA 7.12.2): lane m=lane&15 holds row m; half=lane>>4
//   selects K {half*16 .. half*16+15} (contiguous like B, not the ISA table order).
// B layout (16-bit B KxN): lane holds column N=lane&15, K = (lane>>4)*16 + 0..15
//   -> 16 consecutive bf16 along K from row (jBase+N) of N  (32B contiguous load).
// C/D layout: VGPR r -> M = 8*(lane>>4)+r, N = lane&15.
// ---------------------------------------------------------------------------
__global__ void k_gram_wmma(const bf16_t* __restrict__ nbf, float* __restrict__ sim) {
    const int wave = threadIdx.x >> 5;
    const int lane = threadIdx.x & 31;
    const int tile = blockIdx.x * 8 + wave;       // 0..16383
    const int ti = tile >> 7;
    const int tj = tile & 127;
    const int iBase = ti << 4;
    const int jBase = tj << 4;
    const int m    = lane & 15;
    const int half = lane >> 4;

    const bf16_t* arow = nbf + (size_t)(iBase + m) * D_N;            // A: row m of tile
    const bf16_t* brow = nbf + (size_t)(jBase + m) * D_N + half * 16; // B: col m, K-half

    v8f c = {};
#pragma unroll 4
    for (int k0 = 0; k0 < D_N; k0 += 32) {
        v16bf a = *(const v16bf*)(arow + k0 + half * 16);
        v16bf b = *(const v16bf*)(brow + k0);
        c = __builtin_amdgcn_wmma_f32_16x16x32_bf16(
                /*neg_a=*/false, a, /*neg_b=*/false, b,
                /*c_mod=*/(short)0, c, /*reuse_a=*/false, /*reuse_b=*/false);
    }
    float* out = sim + (size_t)(iBase + 8 * half) * B_N + jBase + m;
#pragma unroll
    for (int r = 0; r < 8; ++r) out[(size_t)r * B_N] = c[r] * TEMP_INV;
}

// ---------------------------------------------------------------------------
// K3: per-row logsumexp (j != i) + positive-pair sums -> row loss / valid
// ---------------------------------------------------------------------------
__global__ void k_row_reduce(const float* __restrict__ sim, const int* __restrict__ labels,
                             float* __restrict__ rowLoss, float* __restrict__ rowValid) {
    __shared__ float smax[256], ssum[256], spos[256];
    __shared__ int   scnt[256];
    const int i = blockIdx.x;
    const int t = threadIdx.x;
    const int li = labels[i];
    const float* row = sim + (size_t)i * B_N;

    float m = -INFINITY, s = 0.f, psum = 0.f;
    int pcnt = 0;
    for (int j = t; j < B_N; j += 256) {
        if (j == i) continue;
        const float v = row[j];
        if (v > m) { s = s * expf(m - v) + 1.0f; m = v; }
        else       { s += expf(v - m); }
        if (labels[j] == li) { psum += v; ++pcnt; }
    }
    smax[t] = m; ssum[t] = s; spos[t] = psum; scnt[t] = pcnt;
    __syncthreads();
    for (int off = 128; off > 0; off >>= 1) {
        if (t < off) {
            float m1 = smax[t], s1 = ssum[t];
            float m2 = smax[t + off], s2 = ssum[t + off];
            float M = fmaxf(m1, m2);
            float S = 0.f;
            if (m1 > -INFINITY) S += s1 * expf(m1 - M);
            if (m2 > -INFINITY) S += s2 * expf(m2 - M);
            smax[t] = M; ssum[t] = S;
            spos[t] += spos[t + off];
            scnt[t] += scnt[t + off];
        }
        __syncthreads();
    }
    if (t == 0) {
        const float log_denom = smax[0] + logf(ssum[0]);
        const int np = scnt[0];
        const float mean_pos = spos[0] / (float)max(np, 1);
        rowLoss[i]  = (np > 0) ? (log_denom - mean_pos) : 0.0f;
        rowValid[i] = (np > 0) ? 1.0f : 0.0f;
    }
}

// ---------------------------------------------------------------------------
// K4: masked squared-error partials over [2048,128,512] (float4 streaming)
// ---------------------------------------------------------------------------
__global__ void k_recon_partial(const float* __restrict__ pred, const float* __restrict__ tgt,
                                const float* __restrict__ mask, float* __restrict__ part) {
    __shared__ float sm[256];
    const size_t nvec = (size_t)B_N * S_N * H_N / 4;   // 33,554,432 float4
    const size_t stride = (size_t)gridDim.x * blockDim.x;
    size_t idx = (size_t)blockIdx.x * blockDim.x + threadIdx.x;
    const float4* p4 = (const float4*)pred;
    const float4* t4 = (const float4*)tgt;

    float acc = 0.f;
    for (; idx < nvec; idx += stride) {
        if (idx + stride < nvec) {   // stream ahead into GL2
            __builtin_prefetch(p4 + idx + stride, 0, 1);
            __builtin_prefetch(t4 + idx + stride, 0, 1);
        }
        const float4 p = p4[idx];
        const float4 q = t4[idx];
        const float w = mask[(idx * 4) >> 9];          // (b*S+s) = flat/H
        const float d0 = p.x - q.x, d1 = p.y - q.y, d2 = p.z - q.z, d3 = p.w - q.w;
        acc += w * (d0 * d0 + d1 * d1 + d2 * d2 + d3 * d3);
    }
    sm[threadIdx.x] = acc; __syncthreads();
    for (int off = 128; off > 0; off >>= 1) {
        if ((int)threadIdx.x < off) sm[threadIdx.x] += sm[threadIdx.x + off];
        __syncthreads();
    }
    if (threadIdx.x == 0) part[blockIdx.x] = sm[0];
}

// ---------------------------------------------------------------------------
// K5: mask-sum partials over [2048,128]
// ---------------------------------------------------------------------------
__global__ void k_mask_partial(const float* __restrict__ mask, float* __restrict__ part) {
    __shared__ float sm[256];
    const int n = B_N * S_N;
    float acc = 0.f;
    for (int i = blockIdx.x * blockDim.x + threadIdx.x; i < n; i += gridDim.x * blockDim.x)
        acc += mask[i];
    sm[threadIdx.x] = acc; __syncthreads();
    for (int off = 128; off > 0; off >>= 1) {
        if ((int)threadIdx.x < off) sm[threadIdx.x] += sm[threadIdx.x + off];
        __syncthreads();
    }
    if (threadIdx.x == 0) part[blockIdx.x] = sm[0];
}

// ---------------------------------------------------------------------------
// K6: finalize -> d_out = {total, contrast, recon}
// ---------------------------------------------------------------------------
__global__ void k_finalize(const float* __restrict__ rowLoss, const float* __restrict__ rowValid,
                           const float* __restrict__ errPart, const float* __restrict__ maskPart,
                           float* __restrict__ out) {
    __shared__ float sL[256], sV[256], sE[256], sM[256];
    const int t = threadIdx.x;
    float l = 0.f, v = 0.f, e = 0.f, mm = 0.f;
    for (int i = t; i < B_N; i += 256) { l += rowLoss[i]; v += rowValid[i]; }
    for (int i = t; i < NB_ERR; i += 256) e += errPart[i];
    if (t < NB_MASK) mm = maskPart[t];
    sL[t] = l; sV[t] = v; sE[t] = e; sM[t] = mm;
    __syncthreads();
    for (int off = 128; off > 0; off >>= 1) {
        if (t < off) { sL[t] += sL[t+off]; sV[t] += sV[t+off]; sE[t] += sE[t+off]; sM[t] += sM[t+off]; }
        __syncthreads();
    }
    if (t == 0) {
        const float contrast = sL[0] / fmaxf(sV[0], 1.0f);
        const float recon = (sE[0] / (float)H_N) / fmaxf(sM[0], 1.0f);
        const float total = GAMMA * contrast + (1.0f - GAMMA) * recon;
        out[0] = total; out[1] = contrast; out[2] = recon;
    }
}

extern "C" void kernel_launch(void* const* d_in, const int* in_sizes, int n_in,
                              void* d_out, int out_size, void* d_ws, size_t ws_size,
                              hipStream_t stream) {
    (void)in_sizes; (void)n_in; (void)out_size; (void)ws_size;
    const float* latents = (const float*)d_in[0];
    const float* recon   = (const float*)d_in[1];
    const float* target  = (const float*)d_in[2];
    const float* mask    = (const float*)d_in[3];
    const int*   labels  = (const int*)d_in[4];
    float* out = (float*)d_out;

    char* ws = (char*)d_ws;
    bf16_t* nbf      = (bf16_t*)(ws + WS_NBF);
    float*  sim      = (float*)(ws + WS_SIM);
    float*  rowLoss  = (float*)(ws + WS_ROWLOSS);
    float*  rowValid = (float*)(ws + WS_ROWVALID);
    float*  errPart  = (float*)(ws + WS_ERRPART);
    float*  maskPart = (float*)(ws + WS_MASKPART);

    k_normalize_bf16<<<B_N, 256, 0, stream>>>(latents, nbf);
    k_gram_wmma<<<(128 * 128) / 8, 256, 0, stream>>>(nbf, sim);
    k_row_reduce<<<B_N, 256, 0, stream>>>(sim, labels, rowLoss, rowValid);
    k_recon_partial<<<NB_ERR, 256, 0, stream>>>(recon, target, mask, errPart);
    k_mask_partial<<<NB_MASK, 256, 0, stream>>>(mask, maskPart);
    k_finalize<<<1, 256, 0, stream>>>(rowLoss, rowValid, errPart, maskPart, out);
}
